// KerasCrossAttentionModule_70334384439646
// MI455X (gfx1250) — compile-verified, matrix operand order checked by reading, not run on hardware
//
#include <hip/hip_runtime.h>
#include <hip/hip_bf16.h>

// ---------------------------------------------------------------------------
// Problem constants (from the reference)
// ---------------------------------------------------------------------------
constexpr int BB = 8;     // batch
constexpr int SS = 4096;  // spatial (64*64)
constexpr int DD = 256;   // head dim (DQK == DV)
constexpr int MT = 128;   // q rows per workgroup (16 per wave, 8 waves)
constexpr int NT = 64;    // keys per tile

typedef __attribute__((ext_vector_type(16))) __bf16 v16bf;
typedef __attribute__((ext_vector_type(8)))  float  v8f;
typedef __attribute__((vector_size(16)))     int    v4i_async;

union FragB {
    v16bf v;
    uint4 u[2];
};

#define GLOBAL_AS __attribute__((address_space(1)))
#define LDS_AS    __attribute__((address_space(3)))

// 16-byte global -> LDS copy. Prefer the CDNA5 async-to-LDS path (no VGPR
// staging, tracked by ASYNCcnt); fall back to a synchronous copy otherwise.
__device__ __forceinline__ void async_copy16(const __bf16* g, __bf16* l) {
#if __has_builtin(__builtin_amdgcn_global_load_async_to_lds_b128)
    __builtin_amdgcn_global_load_async_to_lds_b128(
        (GLOBAL_AS v4i_async*)g, (LDS_AS v4i_async*)l, 0, 0);
#else
    *(uint4*)l = *(const uint4*)g;
#endif
}

__device__ __forceinline__ void wait_async_all() {
    asm volatile("s_wait_asynccnt 0" ::: "memory");
}

// ---------------------------------------------------------------------------
// Prep: (B,D,S) fp32 + pos(S,D) fp32  ->  (B,S,D) bf16   (LDS tile transpose)
// grid (SS/64, DD/64, BB), block 256
// ---------------------------------------------------------------------------
__global__ void prep_addpos_transpose(const float* __restrict__ x,
                                      const float* __restrict__ pos,
                                      __bf16* __restrict__ out) {
    __shared__ float tile[64][65];
    const int s0 = blockIdx.x * 64;
    const int d0 = blockIdx.y * 64;
    const int b  = blockIdx.z;
    const int t  = threadIdx.x;

    const float* xb = x + (size_t)b * DD * SS;
    // phase 1: coalesced reads along s
    {
        const int c  = t & 63;       // s
        const int r4 = t >> 6;       // 0..3
        #pragma unroll
        for (int i = 0; i < 16; ++i) {
            const int r = r4 * 16 + i;   // d
            tile[r][c] = xb[(size_t)(d0 + r) * SS + s0 + c];
        }
    }
    __syncthreads();
    // phase 2: coalesced pos reads + bf16 writes along d
    {
        const int cd  = t & 63;      // d
        const int rs4 = t >> 6;
        #pragma unroll
        for (int i = 0; i < 16; ++i) {
            const int rs = rs4 * 16 + i; // s
            const float v = tile[cd][rs] + pos[(size_t)(s0 + rs) * DD + d0 + cd];
            out[((size_t)b * SS + s0 + rs) * DD + d0 + cd] = (__bf16)v;
        }
    }
}

// ---------------------------------------------------------------------------
// Prep: plain fp32 -> bf16 convert (same layout), vectorized x4
// ---------------------------------------------------------------------------
__global__ void prep_convert(const float* __restrict__ x, __bf16* __restrict__ out, long n) {
    const long i = ((long)blockIdx.x * blockDim.x + threadIdx.x) * 4;
    if (i < n) {
        const float4 f = *(const float4*)(x + i);
        union { __bf16 h[4]; uint2 u; } pk;
        pk.h[0] = (__bf16)f.x;
        pk.h[1] = (__bf16)f.y;
        pk.h[2] = (__bf16)f.z;
        pk.h[3] = (__bf16)f.w;
        *(uint2*)(out + i) = pk.u;
    }
}

// ---------------------------------------------------------------------------
// Flash attention.
//  Qbf, Kbf : (B, S, D) bf16 row-major by token (pos already added)
//  Vbf      : (B, D, S) bf16
//  out      : (B, D, S) fp32
// grid (SS/MT, BB), block 256 (8 wave32)
// ---------------------------------------------------------------------------
__global__ __launch_bounds__(256) void attn_kernel(const __bf16* __restrict__ Qbf,
                                                   const __bf16* __restrict__ Kbf,
                                                   const __bf16* __restrict__ Vbf,
                                                   float* __restrict__ out) {
    __shared__ __bf16 Qs[MT][264];         // q rows, padded (16B-aligned rows)
    __shared__ __bf16 Ks[2][NT][264];      // double-buffered key rows
    __shared__ __bf16 Vs[2][DD][72];       // double-buffered [d][key]
    __shared__ __bf16 Ps[8][16][72];       // per-wave P staging (C- -> A-layout)

    const int tid  = threadIdx.x;
    const int wave = tid >> 5;
    const int lane = tid & 31;
    const int lm   = lane & 15;
    const int hi   = lane >> 4;            // 0: K{0..7,16..23}, 1: K{8..15,24..31}
    const int b    = blockIdx.y;
    const int q0   = blockIdx.x * MT;

    // ---- stage Q tile (async, contiguous 16B chunks) ----
    {
        const __bf16* src = Qbf + ((size_t)b * SS + q0) * DD;
        #pragma unroll
        for (int i = 0; i < 16; ++i) {
            const int u = tid + i * 256;
            const int r = u >> 5, c = u & 31;
            async_copy16(src + (size_t)r * DD + c * 8, &Qs[r][c * 8]);
        }
    }

    // ---- stage the first K/V tile into buffer 0 ----
    const __bf16* kbase = Kbf + (size_t)b * SS * DD;
    const __bf16* vbase = Vbf + (size_t)b * DD * SS;
    {
        #pragma unroll
        for (int i = 0; i < 8; ++i) {
            const int u = tid + i * 256;
            const int r = u >> 5, c = u & 31;
            async_copy16(kbase + (size_t)r * DD + c * 8, &Ks[0][r][c * 8]);
        }
        #pragma unroll
        for (int i = 0; i < 8; ++i) {
            const int u = tid + i * 256;
            const int d = u >> 3, c = u & 7;
            async_copy16(vbase + (size_t)d * SS + c * 8, &Vs[0][d][c * 8]);
        }
    }
    wait_async_all();
    __syncthreads();

    v8f accO[16];
    {
        v8f z = {0.f, 0.f, 0.f, 0.f, 0.f, 0.f, 0.f, 0.f};
        #pragma unroll
        for (int i = 0; i < 16; ++i) accO[i] = z;
    }
    float m_run[8], l_run[8];
    #pragma unroll
    for (int r = 0; r < 8; ++r) { m_run[r] = -1e30f; l_run[r] = 0.f; }

    const float sl = 0.0625f * 1.44269504088896340736f;  // DQK^-0.5 * log2(e)
    const int qrow = wave * 16 + lm;

    for (int j = 0; j < SS / NT; ++j) {
        const int cur = j & 1;

        // ---- prefetch the next K/V tile into the other buffer (async) ----
        if (j + 1 < SS / NT) {
            const int nxt = cur ^ 1;
            const int sk1 = (j + 1) * NT;
            #pragma unroll
            for (int i = 0; i < 8; ++i) {
                const int u = tid + i * 256;
                const int r = u >> 5, c = u & 31;
                async_copy16(kbase + ((size_t)(sk1 + r)) * DD + c * 8, &Ks[nxt][r][c * 8]);
            }
            #pragma unroll
            for (int i = 0; i < 8; ++i) {
                const int u = tid + i * 256;
                const int d = u >> 3, c = u & 7;
                async_copy16(vbase + (size_t)d * SS + sk1 + c * 8, &Vs[nxt][d][c * 8]);
            }
        }

        // ---- scores: S = Q K^T (16 x 64 per wave) ----
        v8f accS[4];
        {
            v8f z = {0.f, 0.f, 0.f, 0.f, 0.f, 0.f, 0.f, 0.f};
            accS[0] = z; accS[1] = z; accS[2] = z; accS[3] = z;
        }
        #pragma unroll
        for (int kc = 0; kc < 8; ++kc) {
            const int db = kc * 32 + hi * 8;
            FragB qa;
            qa.u[0] = *(const uint4*)&Qs[qrow][db];
            qa.u[1] = *(const uint4*)&Qs[qrow][db + 16];
            #pragma unroll
            for (int nk = 0; nk < 4; ++nk) {
                FragB kb;
                kb.u[0] = *(const uint4*)&Ks[cur][nk * 16 + lm][db];
                kb.u[1] = *(const uint4*)&Ks[cur][nk * 16 + lm][db + 16];
                accS[nk] = __builtin_amdgcn_wmma_f32_16x16x32_bf16(
                    false, qa.v, false, kb.v, (short)0, accS[nk], false, false);
            }
        }

        // ---- online softmax (log2 domain) ----
        float mx[8];
        #pragma unroll
        for (int r = 0; r < 8; ++r) {
            float m0 = fmaxf(fmaxf(accS[0][r], accS[1][r]), fmaxf(accS[2][r], accS[3][r]));
            m0 = fmaxf(m0, __shfl_xor(m0, 8, 32));
            m0 = fmaxf(m0, __shfl_xor(m0, 4, 32));
            m0 = fmaxf(m0, __shfl_xor(m0, 2, 32));
            m0 = fmaxf(m0, __shfl_xor(m0, 1, 32));
            mx[r] = m0 * sl;
        }
        float alpha[8], rsum[8];
        #pragma unroll
        for (int r = 0; r < 8; ++r) {
            const float mn = fmaxf(m_run[r], mx[r]);
            alpha[r] = __builtin_amdgcn_exp2f(m_run[r] - mn);
            m_run[r] = mn;
            rsum[r] = 0.f;
        }
        #pragma unroll
        for (int nk = 0; nk < 4; ++nk) {
            #pragma unroll
            for (int r = 0; r < 8; ++r) {
                const float p = __builtin_amdgcn_exp2f(accS[nk][r] * sl - m_run[r]);
                rsum[r] += p;
                Ps[wave][r + hi * 8][nk * 16 + lm] = (__bf16)p;
            }
        }
        #pragma unroll
        for (int r = 0; r < 8; ++r) {
            float rs = rsum[r];
            rs += __shfl_xor(rs, 8, 32);
            rs += __shfl_xor(rs, 4, 32);
            rs += __shfl_xor(rs, 2, 32);
            rs += __shfl_xor(rs, 1, 32);
            l_run[r] = l_run[r] * alpha[r] + rs;
        }
        // rescale O by alpha (rows r / r+8 match the per-lane alpha[r])
        v8f av;
        #pragma unroll
        for (int r = 0; r < 8; ++r) av[r] = alpha[r];
        #pragma unroll
        for (int nd = 0; nd < 16; ++nd) accO[nd] *= av;

        // per-wave LDS P write -> read (type punned): order + wait
        asm volatile("s_wait_dscnt 0" ::: "memory");

        // ---- O += P V ----
        #pragma unroll
        for (int kc2 = 0; kc2 < 2; ++kc2) {
            const int kb2 = kc2 * 32 + hi * 8;
            FragB pa;
            pa.u[0] = *(const uint4*)&Ps[wave][lm][kb2];
            pa.u[1] = *(const uint4*)&Ps[wave][lm][kb2 + 16];
            #pragma unroll
            for (int nd = 0; nd < 16; ++nd) {
                FragB vb;
                vb.u[0] = *(const uint4*)&Vs[cur][nd * 16 + lm][kb2];
                vb.u[1] = *(const uint4*)&Vs[cur][nd * 16 + lm][kb2 + 16];
                accO[nd] = __builtin_amdgcn_wmma_f32_16x16x32_bf16(
                    false, pa.v, false, vb.v, (short)0, accO[nd], false, false);
            }
        }

        // next buffer ready + everyone done reading this one
        wait_async_all();
        __syncthreads();
    }

    // ---- epilogue: divide by l, store (B, D, S) fp32, float4-vectorized ----
    float inv[8];
    #pragma unroll
    for (int r = 0; r < 8; ++r) inv[r] = 1.0f / l_run[r];
    const int srow = q0 + wave * 16 + hi * 8;  // 8 consecutive s per lane
    #pragma unroll
    for (int nd = 0; nd < 16; ++nd) {
        const size_t basep = ((size_t)b * DD + nd * 16 + lm) * SS + srow;
        float4 o0 = {accO[nd][0] * inv[0], accO[nd][1] * inv[1],
                     accO[nd][2] * inv[2], accO[nd][3] * inv[3]};
        float4 o1 = {accO[nd][4] * inv[4], accO[nd][5] * inv[5],
                     accO[nd][6] * inv[6], accO[nd][7] * inv[7]};
        *(float4*)(out + basep)     = o0;
        *(float4*)(out + basep + 4) = o1;
    }
}

// ---------------------------------------------------------------------------
// Host launcher
// ---------------------------------------------------------------------------
extern "C" void kernel_launch(void* const* d_in, const int* in_sizes, int n_in,
                              void* d_out, int out_size, void* d_ws, size_t ws_size,
                              hipStream_t stream) {
    const float* q    = (const float*)d_in[0];
    const float* k    = (const float*)d_in[1];
    const float* v    = (const float*)d_in[2];
    const float* qpos = (const float*)d_in[3];
    const float* kpos = (const float*)d_in[4];
    float* out = (float*)d_out;

    const size_t seg = (size_t)BB * SS * DD * sizeof(__bf16);  // 16 MB
    __bf16* Qbf = (__bf16*)d_ws;
    __bf16* Kbf = (__bf16*)((char*)d_ws + seg);
    __bf16* Vbf = (__bf16*)((char*)d_ws + 2 * seg);

    prep_addpos_transpose<<<dim3(SS / 64, DD / 64, BB), 256, 0, stream>>>(q, qpos, Qbf);
    prep_addpos_transpose<<<dim3(SS / 64, DD / 64, BB), 256, 0, stream>>>(k, kpos, Kbf);
    const long nv = (long)BB * DD * SS;
    prep_convert<<<(int)(nv / 4 / 256), 256, 0, stream>>>(v, Vbf, nv);
    attn_kernel<<<dim3(SS / MT, BB), 256, 0, stream>>>(Qbf, Kbf, Vbf, out);
}
